// Experts_78975858638953
// MI455X (gfx1250) — matrix operand order repaired and checked by reading, not compiled, not run on hardware
//
#include <hip/hip_runtime.h>
#include <math.h>

// ---------------- problem constants (match reference) ----------------
#define NEXP   8
#define DMODEL 768
#define DFFN   3072
#define TTOK   4096           // BATCH*SEQ = 2*2048
#define MTILE  16
#define KSTEP  64             // two 16x16x32 K-slices per LDS stage
#define NTILE  256
#define KPAD   (KSTEP + 8)    // 72 elems = 144B row stride (16B aligned)
#define MT_MAX (TTOK / MTILE + NEXP)   // 264 m-tiles cover worst-case padding

// ---------------- types ----------------
typedef __bf16 bf16;
typedef bf16  v16bf __attribute__((ext_vector_type(16)));
typedef float v8f   __attribute__((ext_vector_type(8)));
typedef unsigned int u32x2 __attribute__((ext_vector_type(2)));
typedef unsigned int u32x4 __attribute__((ext_vector_type(4)));
typedef int          i32x8 __attribute__((ext_vector_type(8)));
typedef float f32x4 __attribute__((ext_vector_type(4)));

union Frag { v16bf v; u32x4 q[2]; };

__device__ __forceinline__ unsigned int pkbf(float a, float b) {
  union { bf16 h[2]; unsigned int u; } p;
  p.h[0] = (bf16)a; p.h[1] = (bf16)b;
  return p.u;                       // lowers to v_cvt_pk_bf16_f32
}

__device__ __forceinline__ float gelu_tanh(float x) {
  const float c0 = 0.7978845608028654f;   // sqrt(2/pi)
  const float c1 = 0.044715f;
  float t = tanhf(c0 * (x + c1 * x * x * x));
  return 0.5f * x * (1.0f + t);
}

// Map a global m-tile index onto (expert, row-base, segment-end).
__device__ __forceinline__ int map_tile(const int* __restrict__ off, int tile,
                                        int* rowbase, int* segend, int* eout) {
  for (int e = 0; e < NEXP; ++e) {
    int s = off[e], t = off[e + 1];
    int nt = (t - s + MTILE - 1) / MTILE;
    if (tile < nt) { *rowbase = s + tile * MTILE; *segend = t; *eout = e; return 1; }
    tile -= nt;
  }
  return 0;
}

// Stage a KSTEP x NTILE fp32 weight tile into LDS as bf16, [n][k] layout.
// 1024 groups of (4k x 4n); 4 groups per thread; ds_store_b64 per column.
__device__ __forceinline__ void stage_b(bf16 (*Bs)[KPAD], const float* __restrict__ wsrc,
                                        int ld, int tid) {
#pragma unroll
  for (int g = 0; g < 4; ++g) {
    int gid = tid + g * 256;
    int kk  = (gid >> 6) * 4;          // 0,4,...,60
    int nn  = (gid & 63) * 4;          // 0,4,...,252
    const float* bp = wsrc + (size_t)kk * ld + nn;
    f32x4 r0 = *(const f32x4*)(bp);
    f32x4 r1 = *(const f32x4*)(bp + ld);
    f32x4 r2 = *(const f32x4*)(bp + 2 * ld);
    f32x4 r3 = *(const f32x4*)(bp + 3 * ld);
    u32x2 d;
    d.x = pkbf(r0.x, r1.x); d.y = pkbf(r2.x, r3.x);
    *(u32x2*)&Bs[nn + 0][kk] = d;
    d.x = pkbf(r0.y, r1.y); d.y = pkbf(r2.y, r3.y);
    *(u32x2*)&Bs[nn + 1][kk] = d;
    d.x = pkbf(r0.z, r1.z); d.y = pkbf(r2.z, r3.z);
    *(u32x2*)&Bs[nn + 2][kk] = d;
    d.x = pkbf(r0.w, r1.w); d.y = pkbf(r2.w, r3.w);
    *(u32x2*)&Bs[nn + 3][kk] = d;
  }
}

// ---------------- kernel 1: routing (counts -> offsets -> permutation) ----------------
__global__ __launch_bounds__(256)
void moe_route(const int* __restrict__ disp, int* __restrict__ off,
               int* __restrict__ perm) {
  __shared__ int cnt[NEXP];
  __shared__ int run[NEXP];
  const int tid = threadIdx.x;
  if (tid < NEXP) cnt[tid] = 0;
  __syncthreads();
  for (int t = tid; t < TTOK; t += 256) atomicAdd(&cnt[disp[t]], 1);
  __syncthreads();
  if (tid == 0) {
    int s = 0;
    for (int e = 0; e < NEXP; ++e) { off[e] = s; run[e] = s; s += cnt[e]; }
    off[NEXP] = s;
  }
  __syncthreads();
  for (int t = tid; t < TTOK; t += 256) {
    int p = atomicAdd(&run[disp[t]], 1);
    perm[p] = t;
  }
}

// ---------------- kernel 2: h = gelu(x[perm] @ W1[e] + b1[e]), bf16 out ----------------
__global__ __launch_bounds__(256)
void moe_gemm1(const float* __restrict__ x, const float* __restrict__ w1,
               const float* __restrict__ bias1, const int* __restrict__ perm,
               const int* __restrict__ off, bf16* __restrict__ h) {
  __shared__ bf16 As[MTILE][KPAD];   // [16][72]
  __shared__ bf16 Bs[NTILE][KPAD];   // [256][72] transposed: [n][k]
  int rowbase, segend, e;
  if (!map_tile(off, blockIdx.x, &rowbase, &segend, &e)) return;

  const int n0   = blockIdx.y * NTILE;
  const int tid  = threadIdx.x;
  const int lane = tid & 31;
  const int wv   = tid >> 5;       // 8 waves
  const int hf   = lane >> 4;      // half-wave select
  const int hm   = lane & 15;

  // per-thread A staging slot: row ar, 4 consecutive k at aseg (loop-invariant gather)
  const int ar   = tid >> 4;
  const int aseg = (tid & 15) * 4;
  const float* arow = x;           // dummy, masked by avalid
  const bool avalid = (rowbase + ar) < segend;
  if (avalid) arow = x + (size_t)perm[rowbase + ar] * DMODEL;

  v8f acc0 = {}; v8f acc1 = {};
  const size_t wbase = (size_t)e * DMODEL * DFFN;

  for (int kb = 0; kb < DMODEL; kb += KSTEP) {
    __syncthreads();
    {   // A tile: 16 tokens x 64 k, gathered, fp32 -> bf16, b64 store
      f32x4 av = {0.f, 0.f, 0.f, 0.f};
      if (avalid) av = *(const f32x4*)(arow + kb + aseg);
      u32x2 d; d.x = pkbf(av.x, av.y); d.y = pkbf(av.z, av.w);
      *(u32x2*)&As[ar][aseg] = d;
    }
    stage_b(Bs, w1 + wbase + (size_t)kb * DFFN + n0, DFFN, tid);
    if (kb + KSTEP < DMODEL)
      __builtin_prefetch(w1 + wbase + (size_t)(kb + KSTEP) * DFFN + n0 + (tid & 63) * 4, 0, 0);
    __syncthreads();

    const int nw = wv * 32;
#pragma unroll
    for (int s = 0; s < 2; ++s) {
      Frag a, b0, b1;
      a.q[0]  = *(const u32x4*)&As[hm][32 * s + 16 * hf];
      a.q[1]  = *(const u32x4*)&As[hm][32 * s + 16 * hf + 8];
      b0.q[0] = *(const u32x4*)&Bs[nw + hm][32 * s + 16 * hf];
      b0.q[1] = *(const u32x4*)&Bs[nw + hm][32 * s + 16 * hf + 8];
      b1.q[0] = *(const u32x4*)&Bs[nw + 16 + hm][32 * s + 16 * hf];
      b1.q[1] = *(const u32x4*)&Bs[nw + 16 + hm][32 * s + 16 * hf + 8];
      acc0 = __builtin_amdgcn_wmma_f32_16x16x32_bf16(false, a.v, false, b0.v, (short)0, acc0, false, false);
      acc1 = __builtin_amdgcn_wmma_f32_16x16x32_bf16(false, a.v, false, b1.v, (short)0, acc1, false, false);
    }
  }

  // epilogue: bias + gelu -> h (bf16)
  const int nc0 = n0 + wv * 32 + hm;
  const int nc1 = nc0 + 16;
  const float g0 = bias1[(size_t)e * DFFN + nc0];
  const float g1 = bias1[(size_t)e * DFFN + nc1];
#pragma unroll
  for (int i = 0; i < 8; ++i) {
    int grow = rowbase + i + 8 * hf;                 // D-matrix: m = i + 8*half
    if (grow < segend) {
      h[(size_t)grow * DFFN + nc0] = (bf16)gelu_tanh(acc0[i] + g0);
      h[(size_t)grow * DFFN + nc1] = (bf16)gelu_tanh(acc1[i] + g1);
    }
  }
}

// ---------------- kernel 3: out[perm] = h @ W2[e] + b2[e], fp32 scatter ----------------
// A tile (bf16, regular 2D) is fetched by the Tensor Data Mover straight into LDS,
// using TDM padding to produce the 144B-padded row stride the fragment loads expect.
__global__ __launch_bounds__(256)
void moe_gemm2(const bf16* __restrict__ h, const float* __restrict__ w2,
               const float* __restrict__ bias2, const int* __restrict__ perm,
               const int* __restrict__ off, float* __restrict__ out) {
  __shared__ bf16 As[MTILE][KPAD];
  __shared__ bf16 Bs[NTILE][KPAD];
  int rowbase, segend, e;
  if (!map_tile(off, blockIdx.x, &rowbase, &segend, &e)) return;

  const int n0   = blockIdx.y * NTILE;
  const int tid  = threadIdx.x;
  const int lane = tid & 31;
  const int wv   = tid >> 5;
  const int hf   = lane >> 4;
  const int hm   = lane & 15;

  v8f acc0 = {}; v8f acc1 = {};
  const size_t wbase = (size_t)e * DFFN * DMODEL;

  // ---- uniform TDM descriptor pieces (D# groups 0/1, ISA 8.3/8.4) ----
  const unsigned int lds_a = (unsigned int)(uintptr_t)&As[0][0];
  const int rows_avail = segend - rowbase;                 // >= 1, fits 16 bits
  const unsigned long long gbase =
      (unsigned long long)(uintptr_t)(h + (size_t)rowbase * DFFN);

  for (int kb = 0; kb < DFFN; kb += KSTEP) {
    __syncthreads();
    if (wv == 0) {   // one wave issues the tensor DMA for the A tile
      unsigned long long ga = gbase + (unsigned long long)kb * 2u;  // byte addr
      u32x4 g0;
      g0.x = 1u;                                              // count=1, user mode
      g0.y = lds_a;                                           // lds_addr (bytes)
      g0.z = (unsigned int)ga;                                // global_addr[31:0]
      g0.w = (unsigned int)((ga >> 32) & 0x1FFFFFFull)        // global_addr[56:32]
             | (2u << 30);                                    // type=2 ("image")
      i32x8 g1;
      g1[0] = (1 << 16)          // data_size = 2 bytes
            | (1 << 20)          // pad_enable
            | (4 << 22)          // pad_interval: 32 DWORDs (=128B, one 64-elem row)
            | (3 << 25);         // pad_amount: 4 DWORDs (=16B -> 144B row stride)
      g1[1] = (int)((DFFN & 0xFFFFu) << 16);                  // tensor_dim0[15:0]
      g1[2] = (int)(((DFFN >> 16) & 0xFFFF)                   // tensor_dim0[31:16]
            | ((rows_avail & 0xFFFF) << 16));                 // tensor_dim1[15:0]
      g1[3] = (KSTEP << 16);                                  // tile_dim0 = 64
      g1[4] = MTILE;                                          // tile_dim1 = 16
      g1[5] = DFFN;                                           // tensor_dim0_stride lo
      g1[6] = 0;
      g1[7] = 0;
      asm volatile("tensor_load_to_lds %0, %1" :: "s"(g0), "s"(g1) : "memory");
    }
    stage_b(Bs, w2 + wbase + (size_t)kb * DMODEL + n0, DMODEL, tid);
    if (kb + KSTEP < DFFN)
      __builtin_prefetch(w2 + wbase + (size_t)(kb + KSTEP) * DMODEL + n0 + (tid & 63) * 4, 0, 0);
    if (wv == 0) __builtin_amdgcn_s_wait_tensorcnt(0);
    __syncthreads();   // publish TDM-written A tile + staged B tile to all waves

    const int nw = wv * 32;
#pragma unroll
    for (int s = 0; s < 2; ++s) {
      Frag a, b0, b1;
      a.q[0]  = *(const u32x4*)&As[hm][32 * s + 16 * hf];
      a.q[1]  = *(const u32x4*)&As[hm][32 * s + 16 * hf + 8];
      b0.q[0] = *(const u32x4*)&Bs[nw + hm][32 * s + 16 * hf];
      b0.q[1] = *(const u32x4*)&Bs[nw + hm][32 * s + 16 * hf + 8];
      b1.q[0] = *(const u32x4*)&Bs[nw + 16 + hm][32 * s + 16 * hf];
      b1.q[1] = *(const u32x4*)&Bs[nw + 16 + hm][32 * s + 16 * hf + 8];
      acc0 = __builtin_amdgcn_wmma_f32_16x16x32_bf16(false, a.v, false, b0.v, (short)0, acc0, false, false);
      acc1 = __builtin_amdgcn_wmma_f32_16x16x32_bf16(false, a.v, false, b1.v, (short)0, acc1, false, false);
    }
  }

  const int nc0 = n0 + wv * 32 + hm;
  const int nc1 = nc0 + 16;
  const float g0 = bias2[(size_t)e * DMODEL + nc0];
  const float g1 = bias2[(size_t)e * DMODEL + nc1];
#pragma unroll
  for (int i = 0; i < 8; ++i) {
    int grow = rowbase + i + 8 * hf;
    if (grow < segend) {
      int orow = perm[grow];
      out[(size_t)orow * DMODEL + nc0] = acc0[i] + g0;
      out[(size_t)orow * DMODEL + nc1] = acc1[i] + g1;
    }
  }
}

// ---------------- launcher ----------------
extern "C" void kernel_launch(void* const* d_in, const int* in_sizes, int n_in,
                              void* d_out, int out_size, void* d_ws, size_t ws_size,
                              hipStream_t stream) {
  const float* x    = (const float*)d_in[0];   // [2,2048,768]
  const int*   disp = (const int*)  d_in[1];   // [4096]
  const float* w1   = (const float*)d_in[2];   // [8,768,3072]
  const float* b1   = (const float*)d_in[3];   // [8,3072]
  const float* w2   = (const float*)d_in[4];   // [8,3072,768]
  const float* b2   = (const float*)d_in[5];   // [8,768]
  float* out = (float*)d_out;

  char* ws = (char*)d_ws;
  int*  off  = (int*)ws;                         // 16 ints   @ 0
  int*  perm = (int*)(ws + 256);                 // 4096 ints @ 256
  bf16* hbuf = (bf16*)(ws + 256 + TTOK * 4);     // 4096*3072 bf16 @ 16640 (128B aligned)

  moe_route<<<1, 256, 0, stream>>>(disp, off, perm);
  moe_gemm1<<<dim3(MT_MAX, DFFN / NTILE), 256, 0, stream>>>(x, w1, b1, perm, off, hbuf);
  moe_gemm2<<<dim3(MT_MAX, DMODEL / NTILE), 256, 0, stream>>>(hbuf, w2, b2, perm, off, out);
}
